// ShapeConnectivityPredictor_88691074662617
// MI455X (gfx1250) — matrix operand order repaired and checked by reading, not compiled, not run on hardware
//
#include <hip/hip_runtime.h>

// ShapeConnectivityPredictor for MI455X (gfx1250, wave32, WMMA).
//
// h_pre[b,i,j] = U[b,i] + V[b,j] + C[b]  (layer-1 factorized: 568 GFLOP -> 7.7 GFLOP)
// Layer-2 GEMM [65536 x 2944] @ [2944 x 1472] ~ 568 GFLOP via v_wmma_f32_16x16x32_bf16,
// h tile LDS-resident, W2 pre-packed bf16 in per-lane WMMA B layout (8.7 MB, L2-resident).
// Layer-3 (mid @ W3) also on WMMA: per pass 2 wmma vs ~88 FMA/thread scalar epilogue.

#define B_GR   64
#define NPER   32
#define D_NODE 320
#define LATENT 512
#define FEAT   1472
#define HID    2944
#define NATOMS 11
#define HPAD   2952   // bf16 row stride for h in LDS: 1476 dwords % 64 banks = 4 -> conflict-free
#define MPAD   72     // bf16 row stride for mid tile: 144 B (16B-aligned, conflict-free)

typedef __attribute__((ext_vector_type(16))) __bf16         v16bf;
typedef __attribute__((ext_vector_type(8)))  float          v8f;
typedef __attribute__((ext_vector_type(4)))  float          v4f;
typedef __attribute__((ext_vector_type(8)))  unsigned short v8us;
typedef __attribute__((ext_vector_type(16))) unsigned short v16us;
typedef __attribute__((ext_vector_type(4)))  unsigned short v4us;

__device__ __forceinline__ unsigned short f2bf(float f) {
  unsigned int u = __builtin_bit_cast(unsigned int, f);
  u += 0x7FFFu + ((u >> 16) & 1u);          // round-to-nearest-even
  return (unsigned short)(u >> 16);
}

__device__ __forceinline__ v8f wmma_bf16(v8us alo, v8us ahi, v8us blo, v8us bhi, v8f acc) {
  v16us av = __builtin_shufflevector(alo, ahi, 0,1,2,3,4,5,6,7,8,9,10,11,12,13,14,15);
  v16us bv = __builtin_shufflevector(blo, bhi, 0,1,2,3,4,5,6,7,8,9,10,11,12,13,14,15);
  return __builtin_amdgcn_wmma_f32_16x16x32_bf16(
      false, __builtin_bit_cast(v16bf, av),
      false, __builtin_bit_cast(v16bf, bv),
      (short)0, acc, false, false);
}

// ---------------- prep kernels ----------------

__global__ void gather_kernel(const int* __restrict__ idx, const int* __restrict__ mlt,
                              const float* __restrict__ id_emb, const float* __restrict__ mu_emb,
                              float* __restrict__ X) {
  int r = blockIdx.x;
  int id = idx[r], mm = mlt[r];
  for (int c = threadIdx.x; c < D_NODE; c += blockDim.x) {
    float v = (c < 256) ? id_emb[id * 256 + c] : mu_emb[mm * 64 + (c - 256)];
    X[r * D_NODE + c] = v;
  }
}

__global__ void agg_kernel(const float* __restrict__ X, float* __restrict__ AGG) {
  int b = blockIdx.x, c = threadIdx.x;
  if (c < D_NODE) {
    float s = 0.f;
    for (int n = 0; n < NPER; ++n) s += X[(b * NPER + n) * D_NODE + c];
    AGG[b * D_NODE + c] = s * (1.0f / NPER);
  }
}

// U = X @ W1[0:320,:], V = X @ W1[320:640,:]
__global__ void uv_kernel(const float* __restrict__ X, const float* __restrict__ W1,
                          float* __restrict__ U, float* __restrict__ V) {
  __shared__ float xs[D_NODE];
  int r = blockIdx.y;
  int h = blockIdx.x * 128 + threadIdx.x;
  for (int f = threadIdx.x; f < D_NODE; f += 128) xs[f] = X[r * D_NODE + f];
  __syncthreads();
  float au = 0.f, av = 0.f;
#pragma unroll 4
  for (int f = 0; f < D_NODE; ++f) {
    float xv = xs[f];
    au = fmaf(xv, W1[(size_t)f * HID + h], au);
    av = fmaf(xv, W1[(size_t)(D_NODE + f) * HID + h], av);
  }
  U[(size_t)r * HID + h] = au;
  V[(size_t)r * HID + h] = av;
}

// C[b] = b1 + z[b] @ W1[640:1152,:] + agg[b] @ W1[1152:1472,:]
__global__ void c_kernel(const float* __restrict__ Z, const float* __restrict__ AGG,
                         const float* __restrict__ W1, const float* __restrict__ b1,
                         float* __restrict__ C) {
  __shared__ float zs[LATENT + D_NODE];
  int b = blockIdx.y;
  int h = blockIdx.x * 128 + threadIdx.x;
  for (int f = threadIdx.x; f < LATENT; f += 128) zs[f] = Z[b * LATENT + f];
  for (int f = threadIdx.x; f < D_NODE; f += 128) zs[LATENT + f] = AGG[b * D_NODE + f];
  __syncthreads();
  float a = b1[h];
#pragma unroll 4
  for (int f = 0; f < LATENT; ++f)
    a = fmaf(zs[f], W1[(size_t)(2 * D_NODE + f) * HID + h], a);
#pragma unroll 4
  for (int f = 0; f < D_NODE; ++f)
    a = fmaf(zs[LATENT + f], W1[(size_t)(2 * D_NODE + LATENT + f) * HID + h], a);
  C[(size_t)b * HID + h] = a;
}

// Pack W2 (f32 [HID][FEAT]) -> bf16 tiles in WMMA B layout.
// Tile (kt,nt) = 32x16, 512 bf16: lane l holds 16 contiguous bf16 =
// column n = nt*16+(l&15), K = kt*32 + (l>>4)*16 + [0..15].
__global__ void packw2_kernel(const float* __restrict__ W2, unsigned short* __restrict__ W2BF) {
  int P = blockIdx.x * 256 + threadIdx.x;
  const int TOT = (HID / 32) * (FEAT / 16) * 512;
  if (P >= TOT) return;
  int kt = P / ((FEAT / 16) * 512);
  int rem = P % ((FEAT / 16) * 512);
  int nt = rem / 512;
  int q = rem % 512;
  int l = q >> 4, kk = q & 15;
  int k = kt * 32 + ((l >> 4) << 4) + kk;
  int n = nt * 16 + (l & 15);
  W2BF[P] = f2bf(W2[(size_t)k * FEAT + n]);
}

// ---------------- fused edge MLP (layers 2+3, all WMMA) ----------------
// One workgroup per (b,i): 32 edges (j = 0..31), 8 waves = 2(M) x 4(N) 16x16 tiles/pass,
// 23 passes over the 1472 columns, K-loop of 92 x wmma_f32_16x16x32_bf16.
// Layer 3: waves 0-1 keep persistent 16x16 logits accumulators, 2 wmma per pass.
__global__ __launch_bounds__(256) void edge_mlp_kernel(
    const float* __restrict__ U, const float* __restrict__ V, const float* __restrict__ C,
    const float* __restrict__ b2, const float* __restrict__ W3, const float* __restrict__ b3,
    const unsigned short* __restrict__ W2BF, float* __restrict__ RAW) {
  __shared__ __align__(16) unsigned short h_sh[NPER * HPAD];    // 184.5 KB bf16 h tile
  __shared__ __align__(16) unsigned short mid_sh[NPER * MPAD];  // 4.5 KB bf16 mid tile (32x64)
  __shared__ __align__(16) unsigned short w3_sh[2 * 512];       // 2 KB: two 32x16 bf16 B tiles

  const int tid = threadIdx.x;
  const int bi  = blockIdx.x;          // b*32 + i
  const int b   = bi >> 5;

  // Stage h[j][k] = relu(U[bi][k] + V[b,j][k] + C[b][k]) as bf16 into LDS.
  {
    int j = tid >> 3, c = tid & 7;
    const float* urow = U + (size_t)bi * HID;
    const float* vrow = V + (size_t)(b * NPER + j) * HID;
    const float* crow = C + (size_t)b * HID;
    for (int kq = c; kq < HID / 4; kq += 8) {
      int k = kq * 4;
      v4f u4 = *(const v4f*)(urow + k);
      v4f v4 = *(const v4f*)(vrow + k);
      v4f c4 = *(const v4f*)(crow + k);
      v4us hv;
#pragma unroll
      for (int t = 0; t < 4; ++t)
        hv[t] = f2bf(fmaxf(u4[t] + v4[t] + c4[t], 0.f));
      *(v4us*)(&h_sh[j * HPAD + k]) = hv;
    }
  }
  __syncthreads();

  const int lane  = tid & 31;
  const int wv    = tid >> 5;
  const int mhalf = wv & 1;            // rows 0-15 or 16-31
  const int nsub  = wv >> 1;           // which 16-col tile of the 64-col pass
  const int mrow  = (lane & 15) + mhalf * 16;   // A-operand row for this lane
  const int kgrp  = (lane >> 4) * 16;           // A-operand K sub-offset (contiguous: K = 16*(lane>>4) + [0..15])
  const int ncol  = lane & 15;                  // C/D column for this lane
  const int mbase = mhalf * 16 + ((lane >> 4) ? 8 : 0);  // C/D row base for this lane

  // Layer-3 persistent accumulators (waves 0 and 1; wave w covers rows w*16..w*16+15)
  v8f lacc = {};
  const int l3row = (lane & 15) + wv * 16;      // A row in mid tile for layer-3 wmma

  const size_t wstride = (size_t)(FEAT / 16) * 512;  // packed-W2 elems per K-tile row

  for (int p = 0; p < FEAT / 64; ++p) {
    __syncthreads();   // protect mid_sh / w3_sh against previous pass's layer-3 reads

    // stage this pass's W3 chunk (64 rows x 11 cols) as two 32x16 bf16 B tiles, cols 11..15 = 0
    for (int t = tid; t < 1024; t += 256) {
      int tile = t >> 9, q = t & 511;
      int l = q >> 4, kk = q & 15;
      int kloc = tile * 32 + ((l >> 4) << 4) + kk;   // 0..63 within pass
      int a    = l & 15;
      float v  = (a < NATOMS) ? W3[(p * 64 + kloc) * NATOMS + a] : 0.f;
      w3_sh[t] = f2bf(v);
    }

    // Layer 2: mid = relu(h @ W2 + b2), one 16x16 tile per wave
    const int ntg = p * 4 + nsub;                 // global 16-column tile index
    v8f acc = {};
    const unsigned short* hbase = h_sh + mrow * HPAD + kgrp;
    const unsigned short* wbase = W2BF + (size_t)ntg * 512 + lane * 16;
#pragma unroll 4
    for (int kt = 0; kt < HID / 32; ++kt) {
      v8us alo = *(const v8us*)(hbase + kt * 32);        // K = kgrp..kgrp+7
      v8us ahi = *(const v8us*)(hbase + kt * 32 + 8);    // K = kgrp+8..kgrp+15
      v8us blo = *(const v8us*)(wbase + kt * wstride);
      v8us bhi = *(const v8us*)(wbase + kt * wstride + 8);
      acc = wmma_bf16(alo, ahi, blo, bhi, acc);
    }
    // bias + ReLU -> bf16 mid tile in LDS (A-operand layout, stride MPAD)
    const float bias = b2[ntg * 16 + ncol];
#pragma unroll
    for (int vr = 0; vr < 8; ++vr)
      mid_sh[(mbase + vr) * MPAD + nsub * 16 + ncol] = f2bf(fmaxf(acc[vr] + bias, 0.f));
    __syncthreads();

    // Layer 3: logits(32x16) += mid(32x64) @ W3chunk(64x16), waves 0-1 only, K=64 -> 2 wmma
    if (wv < 2) {
      const unsigned short* mbaseA = mid_sh + l3row * MPAD + kgrp;
#pragma unroll
      for (int half = 0; half < 2; ++half) {
        v8us alo = *(const v8us*)(mbaseA + half * 32);
        v8us ahi = *(const v8us*)(mbaseA + half * 32 + 8);
        v8us blo = *(const v8us*)(w3_sh + half * 512 + lane * 16);
        v8us bhi = *(const v8us*)(w3_sh + half * 512 + lane * 16 + 8);
        lacc = wmma_bf16(alo, ahi, blo, bhi, lacc);
      }
    }
  }

  // write raw logits (waves 0-1; valid columns a < 11)
  if (wv < 2 && ncol < NATOMS) {
    const float bz = b3[ncol];
#pragma unroll
    for (int vr = 0; vr < 8; ++vr) {
      int m = (wv * 16) + ((lane >> 4) ? 8 : 0) + vr;
      RAW[((size_t)bi * NPER + m) * NATOMS + ncol] = lacc[vr] + bz;
    }
  }
}

// 0.5 * (L[b,i,j] + L[b,j,i])
__global__ void sym_kernel(const float* __restrict__ RAW, float* __restrict__ OUT) {
  int P = blockIdx.x * 256 + threadIdx.x;
  if (P >= B_GR * NPER * NPER * NATOMS) return;
  int e = P / NATOMS, a = P % NATOMS;
  int b = e >> 10, r = e & 1023, i = r >> 5, j = r & 31;
  int e2 = (b << 10) | (j << 5) | i;
  OUT[P] = 0.5f * (RAW[(size_t)e * NATOMS + a] + RAW[(size_t)e2 * NATOMS + a]);
}

// ---------------- launch ----------------

extern "C" void kernel_launch(void* const* d_in, const int* in_sizes, int n_in,
                              void* d_out, int out_size, void* d_ws, size_t ws_size,
                              hipStream_t stream) {
  const int*   idx  = (const int*)d_in[0];
  const int*   mlt  = (const int*)d_in[1];
  const float* Z    = (const float*)d_in[2];
  const float* id_e = (const float*)d_in[3];
  const float* mu_e = (const float*)d_in[4];
  const float* W1   = (const float*)d_in[5];
  const float* b1   = (const float*)d_in[6];
  const float* W2   = (const float*)d_in[7];
  const float* b2   = (const float*)d_in[8];
  const float* W3   = (const float*)d_in[9];
  const float* b3   = (const float*)d_in[10];
  float* out = (float*)d_out;

  // workspace carve (~60.3 MB total)
  float* X   = (float*)d_ws;                          // 2048*320
  float* AGG = X + 2048 * D_NODE;                     // 64*320
  float* U   = AGG + B_GR * D_NODE;                   // 2048*2944
  float* V   = U + (size_t)2048 * HID;                // 2048*2944
  float* C   = V + (size_t)2048 * HID;                // 64*2944
  float* RAW = C + (size_t)B_GR * HID;                // 65536*11
  unsigned short* W2BF = (unsigned short*)(RAW + (size_t)65536 * NATOMS); // 2944*1472 bf16

  gather_kernel<<<B_GR * NPER, 256, 0, stream>>>(idx, mlt, id_e, mu_e, X);
  agg_kernel<<<B_GR, 320, 0, stream>>>(X, AGG);
  uv_kernel<<<dim3(HID / 128, B_GR * NPER), 128, 0, stream>>>(X, W1, U, V);
  c_kernel<<<dim3(HID / 128, B_GR), 128, 0, stream>>>(Z, AGG, W1, b1, C);
  const int totW2 = (HID / 32) * (FEAT / 16) * 512;
  packw2_kernel<<<(totW2 + 255) / 256, 256, 0, stream>>>(W2, W2BF);
  edge_mlp_kernel<<<B_GR * NPER, 256, 0, stream>>>(U, V, C, b2, W3, b3, W2BF, RAW);
  const int totL = B_GR * NPER * NPER * NATOMS;
  sym_kernel<<<(totL + 255) / 256, 256, 0, stream>>>(RAW, out);
}
